// OrthoLinear_75462575391057
// MI455X (gfx1250) — matrix operand order repaired and checked by reading, not compiled, not run on hardware
//
#include <hip/hip_runtime.h>

typedef __attribute__((ext_vector_type(16))) _Float16 v16h;
typedef __attribute__((ext_vector_type(8)))  _Float16 v8h;
typedef __attribute__((ext_vector_type(4)))  _Float16 v4h;
typedef __attribute__((ext_vector_type(8)))  float    v8f;
typedef __attribute__((ext_vector_type(4)))  float    v4f;
typedef __attribute__((ext_vector_type(4)))  unsigned int u32x4;
typedef __attribute__((ext_vector_type(8)))  int      i32x8;
typedef __attribute__((ext_vector_type(4)))  int      i32x4;

#define IN_F   4096
#define OUT_F  4096
#define M_TOT  4096   // 2 * 2048 rows of x
#define ALPHA  1.0f

// GEMM block tile: 128 (M) x 256 (N), K step 32, 8 waves of 64x64 register tiles
#define BM 128
#define BN 256
#define KB 32

// ---------------------------------------------------------------------------
// Kernel 1: W[r, :] = dequant(int4) * scale[r]  (+ CSR residual), f16 output.
// ---------------------------------------------------------------------------
__global__ __launch_bounds__(256) void build_w_kernel(
    const int* __restrict__ packed,     // [OUT_F, IN_F/2], low byte = 2 nibbles
    const float* __restrict__ scales,   // [OUT_F]
    const _Float16* __restrict__ vals,  // [NNZ] fp16
    const int* __restrict__ idx,        // [NNZ]
    const int* __restrict__ ptr,        // [OUT_F+1]
    _Float16* __restrict__ W)           // [OUT_F, IN_F] row-major (K contiguous)
{
    __shared__ __align__(16) _Float16 row[IN_F];
    const int r = blockIdx.x;
    const float s = scales[r];
    const int* prow = packed + (size_t)r * (IN_F / 2);

    for (int j = threadIdx.x; j < IN_F / 2; j += 256) {
        int w = prow[j];
        float lo = ((float)(w & 15) - 8.0f) * s;         // even input col
        float hi = ((float)((w >> 4) & 15) - 8.0f) * s;  // odd input col
        row[2 * j]     = (_Float16)lo;
        row[2 * j + 1] = (_Float16)hi;
    }
    __syncthreads();

    if (threadIdx.x == 0) {   // serial per-row scatter: duplicate cols accumulate safely
        const int b = ptr[r], e = ptr[r + 1];
        for (int j = b; j < e; ++j) {
            const int c = idx[j];
            row[c] = (_Float16)((float)row[c] + ALPHA * (float)vals[j]);
        }
    }
    __syncthreads();

    _Float16* wrow = W + (size_t)r * IN_F;
    for (int j = threadIdx.x; j < IN_F / 8; j += 256)
        ((v8h*)wrow)[j] = ((const v8h*)row)[j];
}

// ---------------------------------------------------------------------------
// Kernel 2: x (f32) -> Xh (f16), vectorized x4
// ---------------------------------------------------------------------------
__global__ __launch_bounds__(256) void cvt_x_kernel(
    const float* __restrict__ x, _Float16* __restrict__ Xh, int n4)
{
    int i = blockIdx.x * 256 + threadIdx.x;
    if (i < n4) {
        v4f v = ((const v4f*)x)[i];
        v4h h;
        #pragma unroll
        for (int k = 0; k < 4; ++k) h[k] = (_Float16)v[k];
        ((v4h*)Xh)[i] = h;
    }
}

// ---------------------------------------------------------------------------
// TDM: issue one 2-D tile load (rows x KB f16 elements, row stride IN_F) from
// global memory into LDS.  D# descriptor packed per CDNA5 ISA ch.8:
//   group0: [1:0]=count=1, [63:32]=lds_addr, [120:64]=global_addr, [127:126]=2
//   group1: data_size=2B, tensor_dim0/1 (OOB), tile_dim0=KB, tile_dim1=rows,
//           tensor_dim0_stride=IN_F, rest 0 (2-D tile, no pad/iterate/mask)
// ---------------------------------------------------------------------------
__device__ __forceinline__ void tdm_load_tile(const _Float16* gtile,
                                              unsigned ldsOff, int rows)
{
    unsigned long long ga = (unsigned long long)(uintptr_t)gtile;
    u32x4 g0;
    g0[0] = 1u;                                              // count=1, user desc
    g0[1] = ldsOff;                                          // LDS byte address
    g0[2] = (unsigned)ga;                                    // global addr lo
    g0[3] = (unsigned)((ga >> 32) & 0x01ffffffULL) | (2u << 30); // addr hi | type=2
    i32x8 g1;
    g1[0] = 1 << 16;                                         // data_size = 2 bytes
    g1[1] = (IN_F & 0xffff) << 16;                           // tensor_dim0 lo16
    g1[2] = ((IN_F >> 16) & 0xffff) | ((OUT_F & 0xffff) << 16); // dim0 hi | dim1 lo
    g1[3] = ((OUT_F >> 16) & 0xffff) | (KB << 16);           // dim1 hi | tile_dim0
    g1[4] = rows & 0xffff;                                   // tile_dim1 (tile_dim2=0)
    g1[5] = IN_F;                                            // dim0_stride lo32
    g1[6] = 0;                                               // stride hi | dim1_stride lo
    g1[7] = 0;
    i32x4 z4 = {0, 0, 0, 0};
#if defined(__clang_major__) && (__clang_major__ >= 23)
    i32x8 z8 = {0, 0, 0, 0, 0, 0, 0, 0};
    __builtin_amdgcn_tensor_load_to_lds(g0, g1, z4, z4, z8, 0);
#else
    __builtin_amdgcn_tensor_load_to_lds(g0, g1, z4, z4, 0);
#endif
}

// ---------------------------------------------------------------------------
// Kernel 3: out = Xh @ W^T via V_WMMA_F32_16X16X32_F16, TDM-fed LDS tiles.
// Block = 8 waves = 2(M) x 4(N); block tile 128x256; wave tile 64x64.
// Double-buffered LDS (48 KB), wave 0 drives the Tensor Data Mover and
// synchronizes with s_wait_tensorcnt + workgroup barrier.
// Fragment layouts (ISA 7.12.2, wave32):
//   A 16x32 f16: lane<16 -> K {0..7,16..23}; lane>=16 -> K {8..15,24..31}
//   B 32x16 f16: lane<16 -> K {0..7,16..23}; lane>=16 -> K {8..15,24..31}, col l16
//   C/D 16x16 f32: vgpr v, lane -> D[M = v + 8*hsel, N = l16]
// ---------------------------------------------------------------------------
__global__ __launch_bounds__(256) void gemm_kernel(
    const _Float16* __restrict__ Xh,   // [M_TOT, IN_F]
    const _Float16* __restrict__ W,    // [OUT_F, IN_F]
    float* __restrict__ out)           // [M_TOT, OUT_F]
{
    __shared__ __align__(16) _Float16 lA[2][BM * KB];   // 2 x 8 KB
    __shared__ __align__(16) _Float16 lB[2][BN * KB];   // 2 x 16 KB

    const int lane = threadIdx.x & 31;
    const int wave = threadIdx.x >> 5;   // 0..7
    const int wm   = wave >> 2;          // 0..1 : M half of block tile
    const int wn   = wave & 3;           // 0..3 : N quarter of block tile
    const int l16  = lane & 15;
    const int hsel = lane >> 4;          // half-wave select

    const int bm = (blockIdx.x & 31) << 7;   // M block base (x128)
    const int bn = (blockIdx.x >> 5) << 8;   // N block base (x256)

    v8f acc[4][4];
    const v8f zero = {0.f, 0.f, 0.f, 0.f, 0.f, 0.f, 0.f, 0.f};
    #pragma unroll
    for (int i = 0; i < 4; ++i)
        #pragma unroll
        for (int j = 0; j < 4; ++j) acc[i][j] = zero;

    // Prologue: wave 0 launches the TDM for K-tile 0 into buffer 0.
    if (wave == 0) {
        tdm_load_tile(Xh + (size_t)bm * IN_F, (unsigned)(uintptr_t)&lA[0][0], BM);
        tdm_load_tile(W  + (size_t)bn * IN_F, (unsigned)(uintptr_t)&lB[0][0], BN);
    }

    const int nk = IN_F / KB;   // 128 K-steps
    int cur = 0;
    for (int ks = 0; ks < nk; ++ks) {
        if (wave == 0) __builtin_amdgcn_s_wait_tensorcnt(0);  // tile[cur] landed
        __syncthreads();   // publish tile[cur]; all waves done reading tile[cur^1]

        if (wave == 0 && ks + 1 < nk) {   // prefetch next K tile into other buffer
            const int kb = (ks + 1) * KB;
            tdm_load_tile(Xh + (size_t)bm * IN_F + kb,
                          (unsigned)(uintptr_t)&lA[cur ^ 1][0], BM);
            tdm_load_tile(W  + (size_t)bn * IN_F + kb,
                          (unsigned)(uintptr_t)&lB[cur ^ 1][0], BN);
        }

        v16h a[4], b[4];
        #pragma unroll
        for (int mi = 0; mi < 4; ++mi) {
            const int row = wm * 64 + mi * 16 + l16;
            v8h a0 = *(const v8h*)&lA[cur][row * KB + hsel * 8];
            v8h a1 = *(const v8h*)&lA[cur][row * KB + 16 + hsel * 8];
            #pragma unroll
            for (int t = 0; t < 8; ++t) { a[mi][t] = a0[t]; a[mi][t + 8] = a1[t]; }
        }
        #pragma unroll
        for (int ni = 0; ni < 4; ++ni) {
            const int row = wn * 64 + ni * 16 + l16;
            v8h b0 = *(const v8h*)&lB[cur][row * KB + hsel * 8];
            v8h b1 = *(const v8h*)&lB[cur][row * KB + 16 + hsel * 8];
            #pragma unroll
            for (int t = 0; t < 8; ++t) { b[ni][t] = b0[t]; b[ni][t + 8] = b1[t]; }
        }

        #pragma unroll
        for (int mi = 0; mi < 4; ++mi)
            #pragma unroll
            for (int ni = 0; ni < 4; ++ni)
                acc[mi][ni] = __builtin_amdgcn_wmma_f32_16x16x32_f16(
                    false, a[mi], false, b[ni],
                    (short)0, acc[mi][ni], false, false);

        cur ^= 1;
    }

    // Write back D tiles (f32)
    #pragma unroll
    for (int mi = 0; mi < 4; ++mi) {
        #pragma unroll
        for (int ni = 0; ni < 4; ++ni) {
            const int col = bn + wn * 64 + ni * 16 + l16;
            #pragma unroll
            for (int v = 0; v < 8; ++v) {
                const int rowi = bm + wm * 64 + mi * 16 + hsel * 8 + v;
                out[(size_t)rowi * OUT_F + col] = acc[mi][ni][v];
            }
        }
    }
}

// ---------------------------------------------------------------------------
extern "C" void kernel_launch(void* const* d_in, const int* in_sizes, int n_in,
                              void* d_out, int out_size, void* d_ws, size_t ws_size,
                              hipStream_t stream)
{
    const float*    x      = (const float*)d_in[0];      // [2,2048,4096] f32
    const int*      packed = (const int*)d_in[1];        // [4096,2048] i32
    const float*    scales = (const float*)d_in[2];      // [4096] f32
    const _Float16* vals   = (const _Float16*)d_in[3];   // [NNZ] f16
    const int*      idx    = (const int*)d_in[4];        // [NNZ] i32
    const int*      ptr    = (const int*)d_in[5];        // [4097] i32
    float* out = (float*)d_out;

    _Float16* Wq = (_Float16*)d_ws;                                        // 32 MB
    _Float16* Xh = (_Float16*)((char*)d_ws + (size_t)OUT_F * IN_F * 2);    // 32 MB

    // 1) dequant + CSR residual -> f16 W
    build_w_kernel<<<OUT_F, 256, 0, stream>>>(packed, scales, vals, idx, ptr, Wq);

    // 2) x f32 -> f16
    const int n4 = (M_TOT * IN_F) / 4;
    cvt_x_kernel<<<(n4 + 255) / 256, 256, 0, stream>>>(x, Xh, n4);

    // 3) TDM-fed WMMA GEMM: 32 x 16 block tiles of 128x256
    gemm_kernel<<<512, 256, 0, stream>>>(Xh, Wq, out);
}
